// ResidualVQ_37666863186436
// MI455X (gfx1250) — matrix operand order repaired and checked by reading, not compiled, not run on hardware
//
#include <hip/hip_runtime.h>

#define BB 8
#define TT 2048
#define DD 256
#define KK 4
#define CSZ 1024
#define ROWS (BB * TT)            // 16384 rows
#define RPB 32                    // rows per block (2 M-tiles)
#define NTHR 128                  // 4 wave32: 2 (M) x 2 (N)
#define CHC 64                    // codes per LDS chunk
#define NCH (CSZ / CHC)           // 16 chunks per stage
#define DSTR 264                  // padded LDS row stride (elements) - kills bank conflicts

typedef __attribute__((ext_vector_type(16))) __bf16 v16bf;
typedef __attribute__((ext_vector_type(8)))  __bf16 v8bf;
typedef __attribute__((ext_vector_type(4)))  __bf16 v4bf;
typedef __attribute__((ext_vector_type(8)))  float  v8f;
typedef __attribute__((ext_vector_type(4)))  unsigned int u32x4;
typedef __attribute__((ext_vector_type(8)))  int i32x8;
typedef __attribute__((ext_vector_type(4)))  int i32x4;

// LDS layout (dynamic):
#define SM_RESB 0                                   // 32*264*2      = 16896 B
#define SM_BBUF 16896                               // 2*64*264*2    = 67584 B
#define SM_REDV (16896 + 67584)                     // 32*2*4        = 256 B
#define SM_REDI (SM_REDV + 256)                     // 256 B
#define SM_BSEL (SM_REDI + 256)                     // 128 B
#define SM_TOTAL (SM_BSEL + 128)                    // 85120 B (< 320 KB WGP LDS)

// Stage fp32 codebooks as bf16 (WMMA operand) + per-code squared norms.
__global__ void rvq_prep_kernel(const float* __restrict__ cb,
                                __bf16* __restrict__ cbb,
                                float* __restrict__ cnorm) {
  __shared__ float red[256];
  const int code = blockIdx.x;              // 0 .. K*CS-1
  const int d = threadIdx.x;                // 0 .. 255
  const size_t off = (size_t)code * DD + d;
  const float v = cb[off];
  cbb[off] = (__bf16)v;
  red[d] = v * v;
  __syncthreads();
  for (int s = 128; s > 0; s >>= 1) {
    if (d < s) red[d] += red[d + s];
    __syncthreads();
  }
  if (d == 0) cnorm[code] = red[0];
}

#if __has_builtin(__builtin_amdgcn_tensor_load_to_lds) && __has_builtin(__builtin_amdgcn_s_wait_tensorcnt)
#define USE_TDM 1
#else
#define USE_TDM 0
#endif

#if USE_TDM
// DMA one 64x256 bf16 codebook chunk into LDS with 16B padding every 512B row.
__device__ inline void tdm_load_chunk(unsigned lds_byte, const __bf16* gsrc) {
  const unsigned long long ga = (unsigned long long)(uintptr_t)gsrc;
  u32x4 g0;
  g0[0] = 1u;                                          // count=1 valid user D#
  g0[1] = lds_byte;                                    // lds_addr
  g0[2] = (unsigned)(ga & 0xFFFFFFFFu);                // global_addr[31:0] -> bits 95:64
  g0[3] = (unsigned)((ga >> 32) & 0x01FFFFFFu) | (2u << 30);  // addr[56:32] | type=2
  i32x8 g1;
  g1[0] = (1 << 16) | (1 << 20) | (6 << 22) | (3 << 25); // data_size=2B, pad: 128 dw -> +4 dw
  g1[1] = (int)((DD & 0xFFFF) << 16);                  // tensor_dim0 = 256 (low16)
  g1[2] = (int)((CSZ & 0xFFFF) << 16);                 // dim0 hi=0 | tensor_dim1 = 1024
  g1[3] = (int)((DD & 0xFFFF) << 16);                  // dim1 hi=0 | tile_dim0 = 256
  g1[4] = CHC;                                         // tile_dim1 = 64, tile_dim2 = 0
  g1[5] = DD;                                          // tensor_dim0_stride = 256 (low32)
  g1[6] = 0;                                           // stride hi | dim1_stride lo
  g1[7] = 0;
  const i32x4 zz4 = {0, 0, 0, 0};
  const i32x8 zz8 = {0, 0, 0, 0, 0, 0, 0, 0};
  // 6-arg variant (amdgpu-toolchain / clang-23 signature)
  __builtin_amdgcn_tensor_load_to_lds(g0, g1, zz4, zz4, zz8, 0);
}
#endif

__launch_bounds__(NTHR)
__global__ void rvq_main_kernel(const float* __restrict__ z,
                                const float* __restrict__ cb,
                                const __bf16* __restrict__ cbb,
                                const float* __restrict__ cnorm,
                                float* __restrict__ zq_out,
                                int* __restrict__ idx_out) {
  extern __shared__ char smem[];
  __bf16* resb = (__bf16*)(smem + SM_RESB);   // [32][DSTR] bf16 residual (A)
  __bf16* bbuf = (__bf16*)(smem + SM_BBUF);   // 2 x [64][DSTR] codebook chunk (B)
  float*  redv = (float*)(smem + SM_REDV);    // [32][2]
  int*    redi = (int*)(smem + SM_REDI);      // [32][2]
  int*    bsel = (int*)(smem + SM_BSEL);      // [32]

  const int tid  = threadIdx.x;
  const int wave = tid >> 5;
  const int lane = tid & 31;
  const int ln16 = lane & 15;
  const int half = lane >> 4;
  const int mg   = wave >> 1;                 // M group: 0..1
  const int ng   = wave & 1;                  // N group: 0..1
  const int mrow = mg * 16 + ln16;            // A-fragment row for this lane
  const int row0 = blockIdx.x * RPB;

  // row/dim ownership for residual upkeep: 4 threads per row, 64 dims each
  const int ur = tid >> 2;
  const int ud = (tid & 3) * 64;

  { // init: residual = bf16(z)
    const float4* src = (const float4*)(z + (size_t)(row0 + ur) * DD + ud);
#pragma unroll
    for (int i = 0; i < 16; ++i) {
      const float4 f = src[i];
      v4bf p;
      p[0] = (__bf16)f.x; p[1] = (__bf16)f.y;
      p[2] = (__bf16)f.z; p[3] = (__bf16)f.w;
      *(v4bf*)(resb + ur * DSTR + ud + 4 * i) = p;
    }
  }

  for (int stage = 0; stage < KK; ++stage) {
    __syncthreads();   // residual tile (and previous-stage reads) settled

    const __bf16* cbb_s = cbb + (size_t)stage * CSZ * DD;
    const float*  cn_s  = cnorm + stage * CSZ;

    float bestv[8];
    int   besti[8];
#pragma unroll
    for (int j = 0; j < 8; ++j) { bestv[j] = 3.4e38f; besti[j] = 0; }

    // ---- stage codebook chunk 0 into LDS ----
#if USE_TDM
    if (wave == 0) {
      tdm_load_chunk((unsigned)(uintptr_t)(bbuf), cbb_s);
      __builtin_amdgcn_s_wait_tensorcnt(0);
    }
#else
#pragma unroll
    for (int p = 0; p < 16; ++p) {
      const int u = tid + p * NTHR;          // 16B unit id, 0..2047
      const int r = u >> 5;                  // 32 units per 512B row
      const int kk = (u & 31) * 8;
      *(v8bf*)(bbuf + r * DSTR + kk) = *(const v8bf*)(cbb_s + r * DD + kk);
    }
#endif
    __syncthreads();

#pragma unroll 1
    for (int c = 0; c < NCH; ++c) {
      const __bf16* bb = bbuf + (c & 1) * (CHC * DSTR);
      // prefetch next chunk into the other buffer
      if (c + 1 < NCH) {
#if USE_TDM
        if (wave == 0)
          tdm_load_chunk((unsigned)(uintptr_t)(bbuf + ((c + 1) & 1) * (CHC * DSTR)),
                         cbb_s + (size_t)(c + 1) * CHC * DD);
#else
        const __bf16* gsrc = cbb_s + (size_t)(c + 1) * CHC * DD;
        __bf16* dstb = bbuf + ((c + 1) & 1) * (CHC * DSTR);
#pragma unroll
        for (int p = 0; p < 16; ++p) {
          const int u = tid + p * NTHR;
          const int r = u >> 5;
          const int kk = (u & 31) * 8;
          *(v8bf*)(dstb + r * DSTR + kk) = *(const v8bf*)(gsrc + r * DD + kk);
        }
#endif
      }

      // A fragments for this wave's 16 rows (re-read per chunk; barriers bound LICM)
      v16bf afr[8];
#pragma unroll
      for (int s = 0; s < 8; ++s) {
        union { v16bf v; v8bf h[2]; } au;
        const int ab = mrow * DSTR + 32 * s + 8 * half;
        au.h[0] = *(const v8bf*)(resb + ab);
        au.h[1] = *(const v8bf*)(resb + ab + 16);
        afr[s] = au.v;
      }

#pragma unroll
      for (int nt = 0; nt < 2; ++nt) {       // this wave's 32 codes of the chunk
        const int lcode = ng * 32 + nt * 16 + ln16;
        const int col = c * CHC + lcode;
        v16bf bfr[8];
#pragma unroll
        for (int s = 0; s < 8; ++s) {
          union { v16bf v; v8bf h[2]; } bu;
          const int bo = lcode * DSTR + 32 * s + 8 * half;
          bu.h[0] = *(const v8bf*)(bb + bo);
          bu.h[1] = *(const v8bf*)(bb + bo + 16);
          bfr[s] = bu.v;
        }
        const float cn = cn_s[col];
        v8f acc = {};
#pragma unroll
        for (int s = 0; s < 8; ++s)          // K = 256 in 8 x 32 steps
          acc = __builtin_amdgcn_wmma_f32_16x16x32_bf16(
              false, afr[s], false, bfr[s], (short)0, acc, false, false);
#pragma unroll
        for (int j = 0; j < 8; ++j) {        // VGPR j = row (half*8+j), col = ln16
          const float dist = cn - 2.0f * acc[j];
          if (dist < bestv[j]) { bestv[j] = dist; besti[j] = col; }
        }
      }

#if USE_TDM
      if (wave == 0) __builtin_amdgcn_s_wait_tensorcnt(0);
#endif
      __syncthreads();   // next chunk visible; everyone done reading current
    }

    // in-wave argmin across the 16 lanes of each half (xor masks keep halves)
#pragma unroll
    for (int j = 0; j < 8; ++j) {
#pragma unroll
      for (int mask = 8; mask >= 1; mask >>= 1) {
        const float ov = __shfl_xor(bestv[j], mask, 32);
        const int   oi = __shfl_xor(besti[j], mask, 32);
        if (ov < bestv[j] || (ov == bestv[j] && oi < besti[j])) {
          bestv[j] = ov; besti[j] = oi;
        }
      }
      if (ln16 == 0) {
        redv[(mg * 16 + half * 8 + j) * 2 + ng] = bestv[j];
        redi[(mg * 16 + half * 8 + j) * 2 + ng] = besti[j];
      }
    }
    __syncthreads();

    // cross-N-group argmin + emit index
    if (tid < RPB) {
      float bv = redv[tid * 2];
      int   bi = redi[tid * 2];
      const float v1 = redv[tid * 2 + 1];
      const int   c1 = redi[tid * 2 + 1];
      if (v1 < bv || (v1 == bv && c1 < bi)) { bv = v1; bi = c1; }
      bsel[tid] = bi;
      idx_out[(size_t)(row0 + tid) * KK + stage] = bi;
    }
    __syncthreads();

    { // residual update (bf16 operand) + exact fp32 output accumulation
      const int cc = bsel[ur];
      const float4* q4 = (const float4*)(cb + ((size_t)stage * CSZ + cc) * DD + ud);
      float4* zq4 = (float4*)(zq_out + (size_t)(row0 + ur) * DD + ud);
#pragma unroll
      for (int i = 0; i < 16; ++i) {
        const float4 q = q4[i];
        v4bf rb = *(const v4bf*)(resb + ur * DSTR + ud + 4 * i);
        v4bf nb;
        nb[0] = (__bf16)((float)rb[0] - q.x);
        nb[1] = (__bf16)((float)rb[1] - q.y);
        nb[2] = (__bf16)((float)rb[2] - q.z);
        nb[3] = (__bf16)((float)rb[3] - q.w);
        *(v4bf*)(resb + ur * DSTR + ud + 4 * i) = nb;
        if (stage == 0) {
          zq4[i] = q;                       // first stage initializes output
        } else {
          float4 o = zq4[i];
          o.x += q.x; o.y += q.y; o.z += q.z; o.w += q.w;
          zq4[i] = o;                       // z_q_final = sum_k q_k (fp32 exact)
        }
      }
    }
  }
}

extern "C" void kernel_launch(void* const* d_in, const int* in_sizes, int n_in,
                              void* d_out, int out_size, void* d_ws, size_t ws_size,
                              hipStream_t stream) {
  (void)in_sizes; (void)n_in; (void)out_size; (void)ws_size;
  const float* z  = (const float*)d_in[0];   // [B,T,D] fp32
  const float* cb = (const float*)d_in[1];   // [K,CS,D] fp32

  __bf16* cbb  = (__bf16*)d_ws;                                        // 2 MB
  float* cnorm = (float*)((char*)d_ws + (size_t)KK * CSZ * DD * sizeof(__bf16)); // 16 KB

  float* zq = (float*)d_out;                     // [B,T,D] fp32
  int* idx  = (int*)(zq + (size_t)ROWS * DD);    // [B,T,K] int32 (concatenated)

  rvq_prep_kernel<<<KK * CSZ, 256, 0, stream>>>(cb, cbb, cnorm);
  rvq_main_kernel<<<ROWS / RPB, NTHR, SM_TOTAL, stream>>>(z, cb, cbb, cnorm, zq, idx);
}
